// ResidueAtomEncoding_33277406609695
// MI455X (gfx1250) — matrix operand order repaired and checked by reading, not compiled, not run on hardware
//
#include <hip/hip_runtime.h>
#include <math.h>

// ---------------------------------------------------------------------------
// EGNN forward for MI455X (gfx1250), wave32 + v_wmma_f32_16x16x32_bf16.
// Edge kernel: one wave = 32-edge x 128-col tile (2x8 v8f accumulators) so
// every B-fragment LDS fetch feeds two WMMAs. __launch_bounds__(256,1) gives
// the compiler the full VGPR budget (LDS limits occupancy anyway) so the
// 128 accumulator VGPRs stay in registers (no scratch spills).
// ---------------------------------------------------------------------------

typedef __attribute__((ext_vector_type(16))) __bf16 v16bf;
typedef __attribute__((ext_vector_type(8)))  float  v8f;

#define C128 128

__device__ __forceinline__ unsigned short bfbits(float x) {
  __bf16 b = (__bf16)x;
  return __builtin_bit_cast(unsigned short, b);
}
__device__ __forceinline__ unsigned int pack_bf2(float lo, float hi) {
  return (unsigned int)bfbits(lo) | ((unsigned int)bfbits(hi) << 16);
}
__device__ __forceinline__ v8f zero8() {
  v8f z = {0.f, 0.f, 0.f, 0.f, 0.f, 0.f, 0.f, 0.f};
  return z;
}
__device__ __forceinline__ v8f wmma_bf16(v16bf a, v16bf b, v8f c) {
  // D = A(16x32 bf16) * B(32x16 bf16) + C(16x16 f32)
  return __builtin_amdgcn_wmma_f32_16x16x32_bf16(false, a, false, b,
                                                 (short)0, c, false, false);
}
__device__ __forceinline__ float silu_f(float x) { return x / (1.f + __expf(-x)); }

// Stage a K x 128 f32 weight (row-major, K multiple of 32) into LDS as bf16
// pairs laid out in per-lane B-fragment order:
//   dword index = ((kk*8 + t)*32 + lane)*8 + q
//   lane: n = t*16 + (lane&15); k0 = kk*32 + 8*(lane>>4) + 2*q + 8*(q>>2)
__device__ void stage_weight(const float* __restrict__ W, unsigned int* sW,
                             int K, int tid, int nth) {
  const int total = (K >> 5) * 2048;
  for (int i = tid; i < total; i += nth) {
    const int q    = i & 7;
    const int lane = (i >> 3) & 31;
    const int t    = (i >> 8) & 7;
    const int kk   = i >> 11;
    const int n    = t * 16 + (lane & 15);
    const int k0   = kk * 32 + ((lane >> 4) << 3) + (q << 1) + ((q >> 2) << 3);
    sW[i] = pack_bf2(W[(size_t)k0 * C128 + n], W[(size_t)(k0 + 1) * C128 + n]);
  }
}

// B fragment: 8 contiguous dwords per lane, 32B aligned -> 2x ds_load_b128.
__device__ __forceinline__ v16bf load_b_frag(const unsigned int* sW, int kk,
                                             int t, int lane) {
  const uint4* p = (const uint4*)(sW + ((((kk << 3) + t) << 5) + lane) * 8);
  union { v16bf v; uint4 q[2]; } b;
  b.q[0] = p[0];
  b.q[1] = p[1];
  return b.v;
}

// A fragment (16x32 bf16) from a row-major bf16 row pointer; two aligned 16B
// chunks -> 2x b128 loads.
__device__ __forceinline__ v16bf load_a_frag_bf(const __bf16* row, int lane) {
  const int h8 = ((lane >> 4) & 1) << 3;
  union { v16bf v; uint4 q[2]; } a;
  a.q[0] = *(const uint4*)(row + h8);
  a.q[1] = *(const uint4*)(row + 16 + h8);
  return a.v;
}

// Same, but source rows are f32 (wide loads, convert to bf16 in VALU).
__device__ __forceinline__ v16bf load_a_frag_f32(const float* row, int lane) {
  const int h8 = ((lane >> 4) & 1) << 3;
  const float4 f0 = *(const float4*)(row + h8);
  const float4 f1 = *(const float4*)(row + h8 + 4);
  const float4 f2 = *(const float4*)(row + 16 + h8);
  const float4 f3 = *(const float4*)(row + 16 + h8 + 4);
  union { v16bf v; unsigned int u[8]; } a;
  a.u[0] = pack_bf2(f0.x, f0.y); a.u[1] = pack_bf2(f0.z, f0.w);
  a.u[2] = pack_bf2(f1.x, f1.y); a.u[3] = pack_bf2(f1.z, f1.w);
  a.u[4] = pack_bf2(f2.x, f2.y); a.u[5] = pack_bf2(f2.z, f2.w);
  a.u[6] = pack_bf2(f3.x, f3.y); a.u[7] = pack_bf2(f3.z, f3.w);
  return a.v;
}

// ---------------------------------------------------------------------------
// radial[e] = || xyz[src[e]] - xyz[dst[e]] ||^2
// ---------------------------------------------------------------------------
__global__ void radial_kernel(const float* __restrict__ xyz,
                              const int* __restrict__ src,
                              const int* __restrict__ dst,
                              float* __restrict__ radial, int Ee) {
  int e = blockIdx.x * blockDim.x + threadIdx.x;
  if (e < Ee) {
    int s = src[e], d = dst[e];
    float dx = xyz[3 * s + 0] - xyz[3 * d + 0];
    float dy = xyz[3 * s + 1] - xyz[3 * d + 1];
    float dz = xyz[3 * s + 2] - xyz[3 * d + 2];
    radial[e] = dx * dx + dy * dy + dz * dz;
  }
}

// ---------------------------------------------------------------------------
// Embedding: out = GELU(LN(X @ W + bias; gamma, beta))  -> bf16 out [N,128]
// ---------------------------------------------------------------------------
template <int K, bool BF16IN>
__global__ __launch_bounds__(256, 1) void embed_kernel(
    const void* __restrict__ Xv, const float* __restrict__ W,
    const float* __restrict__ bias, const float* __restrict__ gamma,
    const float* __restrict__ beta, __bf16* __restrict__ out, int Nn,
    int ntiles) {
  extern __shared__ unsigned int smem[];
  unsigned int* sW = smem;
  const int tid = threadIdx.x, lane = tid & 31, wave = tid >> 5;
  stage_weight(W, sW, K, tid, blockDim.x);
  __syncthreads();

  const int nlo = lane & 15;
  const int h8 = ((lane >> 4) & 1) << 3;
  float bv[8], gv[8], ev[8];
#pragma unroll
  for (int t = 0; t < 8; ++t) {
    bv[t] = bias[t * 16 + nlo];
    gv[t] = gamma[t * 16 + nlo];
    ev[t] = beta[t * 16 + nlo];
  }
  const int wavesTotal = gridDim.x * (blockDim.x >> 5);
  for (int tile = blockIdx.x * (blockDim.x >> 5) + wave; tile < ntiles;
       tile += wavesTotal) {
    const int base = tile * 16;
    int rm = base + nlo; if (rm > Nn - 1) rm = Nn - 1;
    v8f acc[8];
#pragma unroll
    for (int t = 0; t < 8; ++t) acc[t] = zero8();
#pragma unroll
    for (int kk = 0; kk < (K >> 5); ++kk) {
      v16bf a;
      if (BF16IN) a = load_a_frag_bf((const __bf16*)Xv + (size_t)rm * K + kk * 32, lane);
      else        a = load_a_frag_f32((const float*)Xv + (size_t)rm * K + kk * 32, lane);
#pragma unroll
      for (int t = 0; t < 8; ++t)
        acc[t] = wmma_bf16(a, load_b_frag(sW, kk, t, lane), acc[t]);
    }
#pragma unroll
    for (int t = 0; t < 8; ++t)
#pragma unroll
      for (int r = 0; r < 8; ++r) acc[t][r] += bv[t];
    // LayerNorm per row: row (h8 + r) lives in register r across this 16-lane
    // half (n = t*16 + nlo): reduce over t, then shfl_xor within the half.
#pragma unroll
    for (int r = 0; r < 8; ++r) {
      float s = 0.f, s2 = 0.f;
#pragma unroll
      for (int t = 0; t < 8; ++t) { float x = acc[t][r]; s += x; s2 += x * x; }
#pragma unroll
      for (int m = 1; m <= 8; m <<= 1) {
        s  += __shfl_xor(s, m, 32);
        s2 += __shfl_xor(s2, m, 32);
      }
      const float mean = s * (1.f / 128.f);
      const float var  = s2 * (1.f / 128.f) - mean * mean;
      const float rstd = rsqrtf(var + 1e-5f);
      const int row = base + h8 + r;
      if (row < Nn) {
#pragma unroll
        for (int t = 0; t < 8; ++t) {
          float xn = (acc[t][r] - mean) * rstd * gv[t] + ev[t];
          float y = 0.5f * xn * (1.f + erff(xn * 0.70710678118f)); // exact GELU
          out[(size_t)row * C128 + t * 16 + nlo] = (__bf16)y;
        }
      }
    }
  }
}

// ---------------------------------------------------------------------------
// Edge message: msg = silu(silu([h[src]|h[dst]|radial] @ We1 + be1) @ We2 + be2)
// then h_neigh[dst] += msg (f32 atomics). M=32 per wave in GEMM1 (B reuse x2).
// LDS: sW1 (64KB) | w256+biases (1.5KB) | sW2 (32KB) | 8x4KB bounce
// ---------------------------------------------------------------------------
__global__ __launch_bounds__(256, 1) void edge_kernel(
    const __bf16* __restrict__ h, const float* __restrict__ radial,
    const int* __restrict__ src, const int* __restrict__ dst,
    const float* __restrict__ We1, const float* __restrict__ be1,
    const float* __restrict__ We2, const float* __restrict__ be2,
    float* __restrict__ hn, int Ee, int ntiles32) {
  extern __shared__ unsigned int smem[];
  unsigned int* sW1 = smem;                        // 16384 dwords
  float* sAux       = (float*)(smem + 16384);      // w256 | be1 | be2 (384 f32)
  unsigned int* sW2 = smem + 16384 + 384;          // 8192 dwords
  __bf16* bounce    = (__bf16*)(smem + 16384 + 384 + 8192);

  const int tid = threadIdx.x, lane = tid & 31, wave = tid >> 5;
  stage_weight(We1, sW1, 256, tid, blockDim.x);
  stage_weight(We2, sW2, 128, tid, blockDim.x);
  for (int i = tid; i < C128; i += blockDim.x) {
    sAux[i]            = We1[256 * C128 + i];
    sAux[C128 + i]     = be1[i];
    sAux[2 * C128 + i] = be2[i];
  }
  __syncthreads();

  const int nlo = lane & 15;
  const int h8 = ((lane >> 4) & 1) << 3;
  __bf16* myB = bounce + wave * 16 * C128;

  // tail: rank-1 radial + bias + silu -> bounce -> GEMM2 -> silu -> atomics.
  // Per-lane epilogue constants are fetched from LDS here so they are not
  // live across the WMMA main loop.
  auto edge_tail = [&](v8f* acc, int base16) {
    float wrv[8], b1v[8];
#pragma unroll
    for (int t = 0; t < 8; ++t) {
      wrv[t] = sAux[t * 16 + nlo];
      b1v[t] = sAux[C128 + t * 16 + nlo];
    }
#pragma unroll
    for (int r = 0; r < 8; ++r) {
      int er = base16 + h8 + r; if (er > Ee - 1) er = Ee - 1;
      const float rad = radial[er];
#pragma unroll
      for (int t = 0; t < 8; ++t) {
        float x = acc[t][r] + rad * wrv[t] + b1v[t];
        myB[(h8 + r) * C128 + t * 16 + nlo] = (__bf16)silu_f(x);
      }
    }
    asm volatile("s_wait_dscnt 0x0" ::: "memory"); // wave-local LDS RAW

    v8f acc2[8];
#pragma unroll
    for (int t = 0; t < 8; ++t) acc2[t] = zero8();
#pragma unroll
    for (int kk = 0; kk < 4; ++kk) {
      v16bf a = load_a_frag_bf(myB + nlo * C128 + kk * 32, lane);
#pragma unroll
      for (int t = 0; t < 8; ++t)
        acc2[t] = wmma_bf16(a, load_b_frag(sW2, kk, t, lane), acc2[t]);
    }
    float b2v[8];
#pragma unroll
    for (int t = 0; t < 8; ++t) b2v[t] = sAux[2 * C128 + t * 16 + nlo];
#pragma unroll
    for (int r = 0; r < 8; ++r) {
      const int er = base16 + h8 + r;
      if (er < Ee) {
        float* orow = hn + (size_t)dst[er] * C128;
#pragma unroll
        for (int t = 0; t < 8; ++t)
          atomicAdd(orow + t * 16 + nlo, silu_f(acc2[t][r] + b2v[t]));
      }
    }
  };

  const int wavesTotal = gridDim.x * (blockDim.x >> 5);
  for (int tile = blockIdx.x * (blockDim.x >> 5) + wave; tile < ntiles32;
       tile += wavesTotal) {
    const int base = tile * 32;
    int em0 = base + nlo;      if (em0 > Ee - 1) em0 = Ee - 1;
    int em1 = base + 16 + nlo; if (em1 > Ee - 1) em1 = Ee - 1;
    const __bf16* s0 = h + (size_t)src[em0] * C128;
    const __bf16* d0 = h + (size_t)dst[em0] * C128;
    const __bf16* s1 = h + (size_t)src[em1] * C128;
    const __bf16* d1 = h + (size_t)dst[em1] * C128;

    v8f acc0[8], acc1[8];
#pragma unroll
    for (int t = 0; t < 8; ++t) { acc0[t] = zero8(); acc1[t] = zero8(); }

    // f@We1 = h[src]@We1[0:128] + h[dst]@We1[128:256] + radial (x) We1[256,:]
    // Each B fragment feeds both 16-row sub-tiles (2x LDS reuse).
#pragma unroll
    for (int kk = 0; kk < 4; ++kk) {
      v16bf a0 = load_a_frag_bf(s0 + kk * 32, lane);
      v16bf a1 = load_a_frag_bf(s1 + kk * 32, lane);
#pragma unroll
      for (int t = 0; t < 8; ++t) {
        v16bf b = load_b_frag(sW1, kk, t, lane);
        acc0[t] = wmma_bf16(a0, b, acc0[t]);
        acc1[t] = wmma_bf16(a1, b, acc1[t]);
      }
    }
#pragma unroll
    for (int kk = 0; kk < 4; ++kk) {
      v16bf a0 = load_a_frag_bf(d0 + kk * 32, lane);
      v16bf a1 = load_a_frag_bf(d1 + kk * 32, lane);
#pragma unroll
      for (int t = 0; t < 8; ++t) {
        v16bf b = load_b_frag(sW1, kk + 4, t, lane);
        acc0[t] = wmma_bf16(a0, b, acc0[t]);
        acc1[t] = wmma_bf16(a1, b, acc1[t]);
      }
    }
    edge_tail(acc0, base);
    edge_tail(acc1, base + 16);
  }
}

// ---------------------------------------------------------------------------
// Node update: h = silu([h | h_neigh] @ Wn1 + bn1) @ Wn2 + bn2
// In-place on h (each tile's rows touched only by its own wave).
// ---------------------------------------------------------------------------
__global__ __launch_bounds__(256, 1) void node_kernel(
    __bf16* __restrict__ h, const float* __restrict__ hn,
    const float* __restrict__ Wn1, const float* __restrict__ bn1,
    const float* __restrict__ Wn2, const float* __restrict__ bn2,
    float* __restrict__ outF, int writeF32, int Nn, int ntiles) {
  extern __shared__ unsigned int smem[];
  unsigned int* sW1 = smem;                     // 16384 dwords
  float* sAux       = (float*)(smem + 16384);   // bn1 | bn2 (256 f32)
  unsigned int* sW2 = smem + 16384 + 256;       // 8192 dwords
  __bf16* bounce    = (__bf16*)(smem + 16384 + 256 + 8192);

  const int tid = threadIdx.x, lane = tid & 31, wave = tid >> 5;
  stage_weight(Wn1, sW1, 256, tid, blockDim.x);
  stage_weight(Wn2, sW2, 128, tid, blockDim.x);
  for (int i = tid; i < C128; i += blockDim.x) {
    sAux[i]        = bn1[i];
    sAux[C128 + i] = bn2[i];
  }
  __syncthreads();

  const int nlo = lane & 15;
  const int h8 = ((lane >> 4) & 1) << 3;
  __bf16* myB = bounce + wave * 16 * C128;
  const int wavesTotal = gridDim.x * (blockDim.x >> 5);
  for (int tile = blockIdx.x * (blockDim.x >> 5) + wave; tile < ntiles;
       tile += wavesTotal) {
    const int base = tile * 16;
    int rm = base + nlo; if (rm > Nn - 1) rm = Nn - 1;
    const __bf16* hrow = h  + (size_t)rm * C128;
    const float*  nrow = hn + (size_t)rm * C128;

    v8f acc[8];
#pragma unroll
    for (int t = 0; t < 8; ++t) acc[t] = zero8();
#pragma unroll
    for (int kk = 0; kk < 4; ++kk) {
      v16bf a = load_a_frag_bf(hrow + kk * 32, lane);
#pragma unroll
      for (int t = 0; t < 8; ++t)
        acc[t] = wmma_bf16(a, load_b_frag(sW1, kk, t, lane), acc[t]);
    }
#pragma unroll
    for (int kk = 0; kk < 4; ++kk) {
      v16bf a = load_a_frag_f32(nrow + kk * 32, lane);
#pragma unroll
      for (int t = 0; t < 8; ++t)
        acc[t] = wmma_bf16(a, load_b_frag(sW1, kk + 4, t, lane), acc[t]);
    }
    {
      float b1v[8];
#pragma unroll
      for (int t = 0; t < 8; ++t) b1v[t] = sAux[t * 16 + nlo];
#pragma unroll
      for (int r = 0; r < 8; ++r)
#pragma unroll
        for (int t = 0; t < 8; ++t)
          myB[(h8 + r) * C128 + t * 16 + nlo] = (__bf16)silu_f(acc[t][r] + b1v[t]);
    }
    asm volatile("s_wait_dscnt 0x0" ::: "memory");

    v8f acc2[8];
#pragma unroll
    for (int t = 0; t < 8; ++t) acc2[t] = zero8();
#pragma unroll
    for (int kk = 0; kk < 4; ++kk) {
      v16bf a = load_a_frag_bf(myB + nlo * C128 + kk * 32, lane);
#pragma unroll
      for (int t = 0; t < 8; ++t)
        acc2[t] = wmma_bf16(a, load_b_frag(sW2, kk, t, lane), acc2[t]);
    }
    float b2v[8];
#pragma unroll
    for (int t = 0; t < 8; ++t) b2v[t] = sAux[C128 + t * 16 + nlo];
#pragma unroll
    for (int r = 0; r < 8; ++r) {
      const int row = base + h8 + r;
      if (row < Nn) {
#pragma unroll
        for (int t = 0; t < 8; ++t) {
          const float x = acc2[t][r] + b2v[t]; // no activation on final linear
          if (writeF32) outF[(size_t)row * C128 + t * 16 + nlo] = x;
          else          h[(size_t)row * C128 + t * 16 + nlo] = (__bf16)x;
        }
      }
    }
  }
}

// ---------------------------------------------------------------------------
extern "C" void kernel_launch(void* const* d_in, const int* in_sizes, int n_in,
                              void* d_out, int out_size, void* d_ws,
                              size_t ws_size, hipStream_t stream) {
  const float* node_attr = (const float*)d_in[0];
  const float* xyz       = (const float*)d_in[1];
  const int*   src       = (const int*)d_in[2];
  const int*   dst       = (const int*)d_in[3];
  const float* W_in1 = (const float*)d_in[4];
  const float* b_in1 = (const float*)d_in[5];
  const float* g_in1 = (const float*)d_in[6];
  const float* be_in1= (const float*)d_in[7];
  const float* W_in2 = (const float*)d_in[8];
  const float* b_in2 = (const float*)d_in[9];
  const float* g_in2 = (const float*)d_in[10];
  const float* be_in2= (const float*)d_in[11];
  const float* We1 = (const float*)d_in[12];
  const float* be1 = (const float*)d_in[13];
  const float* We2 = (const float*)d_in[14];
  const float* be2 = (const float*)d_in[15];
  const float* Wn1 = (const float*)d_in[16];
  const float* bn1 = (const float*)d_in[17];
  const float* Wn2 = (const float*)d_in[18];
  const float* bn2 = (const float*)d_in[19];

  const int Nn = in_sizes[0] / 64;
  const int Ee = in_sizes[2];
  const int Lc = in_sizes[12] / (257 * 128);

  // workspace carve-out
  char* ws = (char*)d_ws;
  size_t off = 0;
  auto carve = [&](size_t bytes) {
    char* p = ws + off;
    off += (bytes + 255) & ~(size_t)255;
    return p;
  };
  __bf16* h_bf = (__bf16*)carve((size_t)Nn * C128 * 2);
  __bf16* h0   = (__bf16*)carve((size_t)Nn * C128 * 2);
  float*  hn   = (float*)carve((size_t)Nn * C128 * 4);
  float*  rad  = (float*)carve((size_t)Ee * 4);
  (void)ws_size; (void)n_in; (void)out_size;

  const int ntilesN  = (Nn + 15) / 16;
  const int ntilesE2 = (Ee + 31) / 32;
  int blkN = (ntilesN + 7) / 8;  if (blkN > 4096) blkN = 4096; if (blkN < 1) blkN = 1;
  int blkE = (ntilesE2 + 7) / 8; if (blkE > 2048) blkE = 2048; if (blkE < 1) blkE = 1;

  const size_t shm1 = (64  / 32) * 2048 * 4;                                // 16 KB
  const size_t shm2 = (128 / 32) * 2048 * 4;                                // 32 KB
  const size_t shmE = 16384 * 4 + 384 * 4 + 8192 * 4 + 8 * 16 * C128 * 2;   // ~129 KB
  const size_t shmN = 16384 * 4 + 256 * 4 + 8192 * 4 + 8 * 16 * C128 * 2;   // ~129 KB

  // initial node embedding (f32 in, bf16 out via h0, then h_bf)
  embed_kernel<64, false><<<blkN, 256, shm1, stream>>>(
      (const void*)node_attr, W_in1, b_in1, g_in1, be_in1, h0, Nn, ntilesN);
  embed_kernel<128, true><<<blkN, 256, shm2, stream>>>(
      (const void*)h0, W_in2, b_in2, g_in2, be_in2, h_bf, Nn, ntilesN);

  radial_kernel<<<(Ee + 255) / 256, 256, 0, stream>>>(xyz, src, dst, rad, Ee);

  for (int l = 0; l < Lc; ++l) {
    hipMemsetAsync(hn, 0, (size_t)Nn * C128 * 4, stream);
    edge_kernel<<<blkE, 256, shmE, stream>>>(
        h_bf, rad, src, dst,
        We1 + (size_t)l * 257 * 128, be1 + (size_t)l * 128,
        We2 + (size_t)l * 128 * 128, be2 + (size_t)l * 128, hn, Ee, ntilesE2);
    node_kernel<<<blkN, 256, shmN, stream>>>(
        h_bf, hn,
        Wn1 + (size_t)l * 256 * 128, bn1 + (size_t)l * 128,
        Wn2 + (size_t)l * 128 * 128, bn2 + (size_t)l * 128,
        (float*)d_out, (l == Lc - 1) ? 1 : 0, Nn, ntilesN);
  }
}
